// Encoderlayer_17961553231937
// MI455X (gfx1250) — compile-verified, matrix operand order checked by reading, not run on hardware
//
#include <hip/hip_runtime.h>
#include <hip/hip_bf16.h>
#include <math.h>

// ---------------- problem constants ----------------
constexpr int Hdim  = 1024;
constexpr int Sdim  = 1024;
constexpr int Bdim  = 4;
constexpr int NHEAD = 32;
constexpr int DHEAD = 32;
constexpr int WD    = 512;
constexpr float ATT_SCALE = 0.17677669529663687f;   // 1/sqrt(32)
constexpr float WGAIN     = 0.04419417382415922f;   // 1/sqrt(512)
constexpr int Mtot  = Bdim * Sdim;                  // 4096
constexpr size_t HH = (size_t)Hdim * Hdim;

typedef __bf16 bf16_t;
typedef __attribute__((ext_vector_type(16))) __bf16 v16bf;
typedef __attribute__((ext_vector_type(8)))  float  v8f;

__device__ __forceinline__ v8f wmma_bf16(v16bf a, v16bf b, v8f c) {
  return __builtin_amdgcn_wmma_f32_16x16x32_bf16(false, a, false, b, (short)0, c, false, false);
}

__device__ __forceinline__ float waveRedSum(float v) {
#pragma unroll
  for (int off = 16; off; off >>= 1) v += __shfl_xor(v, off);
  return v;
}

// ---------------- workspace layout (bytes) ----------------
constexpr size_t OFF_STYLES = 0;                                         // B*2H f32
constexpr size_t OFF_SCL    = OFF_STYLES + (size_t)Bdim * 2 * Hdim * 4;  // 5*B*H f32
constexpr size_t OFF_WBF    = OFF_SCL    + (size_t)5 * Bdim * Hdim * 4;  // 5*H*H bf16
constexpr size_t OFF_XN     = OFF_WBF    + 5 * HH * 2;                   // M*H bf16
constexpr size_t OFF_Q      = OFF_XN     + (size_t)Mtot * Hdim * 2;
constexpr size_t OFF_K      = OFF_Q      + (size_t)Mtot * Hdim * 2;
constexpr size_t OFF_V      = OFF_K      + (size_t)Mtot * Hdim * 2;
constexpr size_t OFF_VT     = OFF_V      + (size_t)Mtot * Hdim * 2;      // B,NH,D,S bf16
constexpr size_t OFF_O      = OFF_VT     + (size_t)Mtot * Hdim * 2;      // M*H bf16

// ---------------- K0: styles = w @ (affine_w.T * WGAIN) + affine_b ----------------
__global__ void styles_kernel(const float* __restrict__ w, const float* __restrict__ aw,
                              const float* __restrict__ ab, float* __restrict__ styles) {
  int o = blockIdx.x * blockDim.x + threadIdx.x;  // 0..2H-1
  int b = blockIdx.y;
  const float* wr = w + (size_t)b * WD;
  const float* ar = aw + (size_t)o * WD;
  float acc = 0.f;
  for (int i = 0; i < WD; ++i) acc += wr[i] * ar[i];
  styles[(size_t)b * 2 * Hdim + o] = acc * WGAIN + ab[o];
}

// ---------------- K1: weights -> bf16 ----------------
__global__ void cvt_kernel(const float* __restrict__ qw, const float* __restrict__ kw,
                           const float* __restrict__ vw, const float* __restrict__ ww,
                           const float* __restrict__ uw, bf16_t* __restrict__ wbf) {
  size_t i = (size_t)blockIdx.x * blockDim.x + threadIdx.x;
  wbf[0 * HH + i] = (bf16_t)qw[i];
  wbf[1 * HH + i] = (bf16_t)kw[i];
  wbf[2 * HH + i] = (bf16_t)vw[i];
  wbf[3 * HH + i] = (bf16_t)ww[i];
  wbf[4 * HH + i] = (bf16_t)uw[i];
}

// ---------------- K2: dcoef scales (v fused with s2) ----------------
__global__ void dcoef_kernel(const float* __restrict__ qw, const float* __restrict__ kw,
                             const float* __restrict__ vw, const float* __restrict__ ww,
                             const float* __restrict__ uw, const float* __restrict__ styles,
                             float* __restrict__ scl) {
  int o = blockIdx.x, mat = blockIdx.y;
  const float* W;
  switch (mat) { case 0: W = qw; break; case 1: W = kw; break;
                 case 2: W = vw; break; case 3: W = ww; break; default: W = uw; }
  W += (size_t)o * Hdim;
  int soff = (mat < 3) ? 0 : Hdim;
  float a0 = 0, a1 = 0, a2 = 0, a3 = 0;
  for (int i = threadIdx.x; i < Hdim; i += 256) {
    float wv = W[i];
    float m0 = wv * styles[0 * 2 * Hdim + soff + i]; a0 += m0 * m0;
    float m1 = wv * styles[1 * 2 * Hdim + soff + i]; a1 += m1 * m1;
    float m2 = wv * styles[2 * 2 * Hdim + soff + i]; a2 += m2 * m2;
    float m3 = wv * styles[3 * 2 * Hdim + soff + i]; a3 += m3 * m3;
  }
  a0 = waveRedSum(a0); a1 = waveRedSum(a1); a2 = waveRedSum(a2); a3 = waveRedSum(a3);
  __shared__ float red[8][4];
  int wid = threadIdx.x >> 5, L = threadIdx.x & 31;
  if (!L) { red[wid][0] = a0; red[wid][1] = a1; red[wid][2] = a2; red[wid][3] = a3; }
  __syncthreads();
  if (threadIdx.x < 4) {
    int b = threadIdx.x;
    float s = 0;
    for (int i2 = 0; i2 < 8; ++i2) s += red[i2][b];
    float r = rsqrtf(s + 1e-8f);
    if (mat == 2) r *= styles[(size_t)b * 2 * Hdim + Hdim + o];  // fold s2 into v scale
    scl[((size_t)mat * Bdim + b) * Hdim + o] = r;
  }
}

// ---------------- K3: x*s1 -> layernorm -> bf16 ----------------
__global__ void ln_kernel(const float* __restrict__ x, const float* __restrict__ styles,
                          bf16_t* __restrict__ xn) {
  int row = blockIdx.x;            // b*S + s
  int b   = row >> 10;
  const float* xr = x + (size_t)row * Hdim;
  const float* s1 = styles + (size_t)b * 2 * Hdim;
  float sum = 0.f, sq = 0.f;
  for (int i = threadIdx.x; i < Hdim; i += 256) {
    float z = xr[i] * s1[i];
    sum += z; sq += z * z;
  }
  sum = waveRedSum(sum); sq = waveRedSum(sq);
  __shared__ float rs[8][2];
  int wid = threadIdx.x >> 5, L = threadIdx.x & 31;
  if (!L) { rs[wid][0] = sum; rs[wid][1] = sq; }
  __syncthreads();
  if (threadIdx.x == 0) {
    float a = 0, c = 0;
    for (int i = 0; i < 8; ++i) { a += rs[i][0]; c += rs[i][1]; }
    float mu = a * (1.0f / Hdim);
    rs[0][0] = mu;
    rs[0][1] = rsqrtf(c * (1.0f / Hdim) - mu * mu + 1e-5f);
  }
  __syncthreads();
  float mu = rs[0][0], inv = rs[0][1];
  for (int i = threadIdx.x; i < Hdim; i += 256)
    xn[(size_t)row * Hdim + i] = (bf16_t)((xr[i] * s1[i] - mu) * inv);
}

// ---------------- K4: scaled bf16 WMMA GEMM: out[m,n] = (A @ W^T)[m,n] * scl[b,n] ----------------
__global__ __launch_bounds__(256) void gemm_scaled(const bf16_t* __restrict__ A,
                                                   const bf16_t* __restrict__ W,
                                                   const float* __restrict__ scl,
                                                   bf16_t* __restrict__ out) {
  int wid = threadIdx.x >> 5, L = threadIdx.x & 31, half = L >> 4, l16 = L & 15;
  int m0 = blockIdx.y * 128 + wid * 16;
  int n0 = blockIdx.x * 64;
  v8f acc[4] = {};
  const bf16_t* ap = A + (size_t)(m0 + l16) * Hdim + 16 * half;
  const bf16_t* bp = W + (size_t)(n0 + l16) * Hdim + 16 * half;
  for (int k = 0; k < Hdim; k += 32) {
    v16bf a = *(const v16bf*)(ap + k);
#pragma unroll
    for (int t = 0; t < 4; ++t) {
      v16bf bm = *(const v16bf*)(bp + (size_t)t * 16 * Hdim + k);
      acc[t] = wmma_bf16(a, bm, acc[t]);
    }
  }
  int b = m0 >> 10;
#pragma unroll
  for (int t = 0; t < 4; ++t) {
    int n = n0 + t * 16 + l16;
    float sc = scl[(size_t)b * Hdim + n];
#pragma unroll
    for (int j = 0; j < 8; ++j) {
      int m = m0 + j + 8 * half;
      out[(size_t)m * Hdim + n] = (bf16_t)(acc[t][j] * sc);
    }
  }
}

// ---------------- K5: V head-transpose: vt[b,h,d,s] = v[b,s,h*32+d] ----------------
__global__ void vtrans_kernel(const bf16_t* __restrict__ v, bf16_t* __restrict__ vt) {
  __shared__ bf16_t tile[32][33];
  int b = blockIdx.z, h = blockIdx.y, s0 = blockIdx.x * 32;
  int tx = threadIdx.x, ty = threadIdx.y;   // 32 x 8
#pragma unroll
  for (int i = 0; i < 4; ++i) {
    int sl = ty + 8 * i;
    tile[sl][tx] = v[((size_t)(b * Sdim + s0 + sl)) * Hdim + h * DHEAD + tx];
  }
  __syncthreads();
#pragma unroll
  for (int i = 0; i < 4; ++i) {
    int dl = ty + 8 * i;
    vt[((size_t)((b * NHEAD + h) * DHEAD + dl)) * Sdim + s0 + tx] = tile[tx][dl];
  }
}

// ---------------- K6: flash attention (wave per 16-query tile) ----------------
__global__ __launch_bounds__(256) void attn_kernel(const bf16_t* __restrict__ q,
                                                   const bf16_t* __restrict__ k,
                                                   const bf16_t* __restrict__ vt,
                                                   bf16_t* __restrict__ o) {
  __shared__ __align__(32) bf16_t pst[8][512];   // per-wave 16x32 P staging
  int wid = threadIdx.x >> 5, L = threadIdx.x & 31, half = L >> 4, l16 = L & 15;
  int b = blockIdx.z, h = blockIdx.y;
  int q0 = blockIdx.x * 128 + wid * 16;

  v16bf qa = *(const v16bf*)(q + ((size_t)(b * Sdim + q0 + l16)) * Hdim + h * DHEAD + 16 * half);
  v8f o0 = {}, o1 = {};
  float mrun[8], lrun[8];
#pragma unroll
  for (int j = 0; j < 8; ++j) { mrun[j] = -1e30f; lrun[j] = 0.f; }

  const bf16_t* krow = k  + ((size_t)(b * Sdim)) * Hdim + h * DHEAD + 16 * half;
  const bf16_t* vrow = vt + ((size_t)((b * NHEAD + h) * DHEAD)) * Sdim + 16 * half;

  for (int kb = 0; kb < Sdim; kb += 32) {
    v16bf kb0 = *(const v16bf*)(krow + (size_t)(kb + l16) * Hdim);
    v16bf kb1 = *(const v16bf*)(krow + (size_t)(kb + 16 + l16) * Hdim);
    v8f z = {};
    v8f c0 = wmma_bf16(qa, kb0, z);
    v8f c1 = wmma_bf16(qa, kb1, z);

    float rm[8];
#pragma unroll
    for (int j = 0; j < 8; ++j) {
      c0[j] *= ATT_SCALE; c1[j] *= ATT_SCALE;
      rm[j] = fmaxf(c0[j], c1[j]);
    }
#pragma unroll
    for (int off = 8; off; off >>= 1)
#pragma unroll
      for (int j = 0; j < 8; ++j) rm[j] = fmaxf(rm[j], __shfl_xor(rm[j], off));

    float alpha[8], rsum[8];
#pragma unroll
    for (int j = 0; j < 8; ++j) {
      float mn = fmaxf(mrun[j], rm[j]);
      alpha[j] = __expf(mrun[j] - mn);
      mrun[j] = mn;
      c0[j] = __expf(c0[j] - mn);
      c1[j] = __expf(c1[j] - mn);
      rsum[j] = c0[j] + c1[j];
    }
#pragma unroll
    for (int off = 8; off; off >>= 1)
#pragma unroll
      for (int j = 0; j < 8; ++j) rsum[j] += __shfl_xor(rsum[j], off);
#pragma unroll
    for (int j = 0; j < 8; ++j) {
      lrun[j] = lrun[j] * alpha[j] + rsum[j];
      o0[j] *= alpha[j];
      o1[j] *= alpha[j];
    }

    // C-layout -> A-layout via wave-private LDS tile
#pragma unroll
    for (int j = 0; j < 8; ++j) {
      int r = j + 8 * half;
      pst[wid][r * 32 + l16]      = (bf16_t)c0[j];
      pst[wid][r * 32 + 16 + l16] = (bf16_t)c1[j];
    }
    asm volatile("s_wait_dscnt 0" ::: "memory");
    v16bf pa  = *(const v16bf*)(&pst[wid][l16 * 32 + 16 * half]);
    v16bf vb0 = *(const v16bf*)(vrow + (size_t)l16 * Sdim + kb);
    v16bf vb1 = *(const v16bf*)(vrow + (size_t)(16 + l16) * Sdim + kb);
    o0 = wmma_bf16(pa, vb0, o0);
    o1 = wmma_bf16(pa, vb1, o1);
  }

#pragma unroll
  for (int j = 0; j < 8; ++j) {
    float il = 1.f / lrun[j];
    o0[j] *= il; o1[j] *= il;
  }
  size_t ob = ((size_t)(b * Sdim + q0)) * Hdim + h * DHEAD;
#pragma unroll
  for (int j = 0; j < 8; ++j) {
    int m = j + 8 * half;
    o[ob + (size_t)m * Hdim + l16]      = (bf16_t)o0[j];
    o[ob + (size_t)m * Hdim + 16 + l16] = (bf16_t)o1[j];
  }
}

// ---------------- K7: fused dual GEMM + epilogue (f32 out) ----------------
__global__ __launch_bounds__(256) void final_kernel(const bf16_t* __restrict__ Ao,
                                                    const bf16_t* __restrict__ Av,
                                                    const bf16_t* __restrict__ Ww,
                                                    const bf16_t* __restrict__ Wu,
                                                    const float* __restrict__ wd,
                                                    const float* __restrict__ ud,
                                                    const float* __restrict__ noise,
                                                    const float* __restrict__ ns,
                                                    const float* __restrict__ bias,
                                                    float* __restrict__ out) {
  int wid = threadIdx.x >> 5, L = threadIdx.x & 31, half = L >> 4, l16 = L & 15;
  int m0 = blockIdx.y * 128 + wid * 16;
  int n0 = blockIdx.x * 64;
  v8f acc1[4] = {}, acc2[4] = {};
  const bf16_t* a1p = Ao + (size_t)(m0 + l16) * Hdim + 16 * half;
  const bf16_t* a2p = Av + (size_t)(m0 + l16) * Hdim + 16 * half;
  const bf16_t* b1p = Ww + (size_t)(n0 + l16) * Hdim + 16 * half;
  const bf16_t* b2p = Wu + (size_t)(n0 + l16) * Hdim + 16 * half;
  for (int kk = 0; kk < Hdim; kk += 32) {
    v16bf a1 = *(const v16bf*)(a1p + kk);
    v16bf a2 = *(const v16bf*)(a2p + kk);
#pragma unroll
    for (int t = 0; t < 4; ++t) {
      v16bf bb1 = *(const v16bf*)(b1p + (size_t)t * 16 * Hdim + kk);
      v16bf bb2 = *(const v16bf*)(b2p + (size_t)t * 16 * Hdim + kk);
      acc1[t] = wmma_bf16(a1, bb1, acc1[t]);
      acc2[t] = wmma_bf16(a2, bb2, acc2[t]);
    }
  }
  int b = m0 >> 10;
  float nstr = ns[0];
#pragma unroll
  for (int t = 0; t < 4; ++t) {
    int n = n0 + t * 16 + l16;
    float sw = wd[(size_t)b * Hdim + n];
    float su = ud[(size_t)b * Hdim + n];
    float bv = bias[n];
#pragma unroll
    for (int j = 0; j < 8; ++j) {
      int m = m0 + j + 8 * half;
      float val = acc1[t][j] * sw + acc2[t][j] * su + noise[m] * nstr + bv;
      val = (val >= 0.f) ? val : 0.2f * val;
      val = fminf(fmaxf(val, -256.f), 256.f);
      out[(size_t)m * Hdim + n] = val;
    }
  }
}

// ---------------- launcher ----------------
extern "C" void kernel_launch(void* const* d_in, const int* in_sizes, int n_in,
                              void* d_out, int out_size, void* d_ws, size_t ws_size,
                              hipStream_t stream) {
  (void)in_sizes; (void)n_in; (void)out_size; (void)ws_size;
  const float* x     = (const float*)d_in[0];
  const float* w     = (const float*)d_in[1];
  const float* aw    = (const float*)d_in[2];
  const float* ab    = (const float*)d_in[3];
  const float* qw    = (const float*)d_in[4];
  const float* kw    = (const float*)d_in[5];
  const float* vw    = (const float*)d_in[6];
  const float* wwt   = (const float*)d_in[7];
  const float* uwt   = (const float*)d_in[8];
  const float* bias  = (const float*)d_in[9];
  const float* ns    = (const float*)d_in[10];
  const float* noise = (const float*)d_in[11];
  float* out = (float*)d_out;

  char* ws = (char*)d_ws;
  float*  styles = (float*)(ws + OFF_STYLES);
  float*  scl    = (float*)(ws + OFF_SCL);
  bf16_t* wbf    = (bf16_t*)(ws + OFF_WBF);
  bf16_t* xn     = (bf16_t*)(ws + OFF_XN);
  bf16_t* qb     = (bf16_t*)(ws + OFF_Q);
  bf16_t* kb     = (bf16_t*)(ws + OFF_K);
  bf16_t* vb     = (bf16_t*)(ws + OFF_V);
  bf16_t* vt     = (bf16_t*)(ws + OFF_VT);
  bf16_t* ob     = (bf16_t*)(ws + OFF_O);

  styles_kernel<<<dim3(2 * Hdim / 256, Bdim), 256, 0, stream>>>(w, aw, ab, styles);
  cvt_kernel<<<(int)(HH / 256), 256, 0, stream>>>(qw, kw, vw, wwt, uwt, wbf);
  dcoef_kernel<<<dim3(Hdim, 5), 256, 0, stream>>>(qw, kw, vw, wwt, uwt, styles, scl);
  ln_kernel<<<Mtot, 256, 0, stream>>>(x, styles, xn);

  dim3 ggrid(Hdim / 64, Mtot / 128);
  gemm_scaled<<<ggrid, 256, 0, stream>>>(xn, wbf + 0 * HH, scl + 0 * Bdim * Hdim, qb);
  gemm_scaled<<<ggrid, 256, 0, stream>>>(xn, wbf + 1 * HH, scl + 1 * Bdim * Hdim, kb);
  gemm_scaled<<<ggrid, 256, 0, stream>>>(xn, wbf + 2 * HH, scl + 2 * Bdim * Hdim, vb);

  vtrans_kernel<<<dim3(Sdim / 32, NHEAD, Bdim), dim3(32, 8), 0, stream>>>(vb, vt);
  attn_kernel<<<dim3(Sdim / 128, NHEAD, Bdim), 256, 0, stream>>>(qb, kb, vt, ob);

  final_kernel<<<ggrid, 256, 0, stream>>>(ob, vb, wbf + 3 * HH, wbf + 4 * HH,
                                          scl + 3 * Bdim * Hdim, scl + 4 * Bdim * Hdim,
                                          noise, ns, bias, out);
}
